// GCN_41326175322234
// MI455X (gfx1250) — matrix operand order repaired and checked by reading, not compiled, not run on hardware
//
#include <hip/hip_runtime.h>
#include <hip/hip_bf16.h>

typedef __attribute__((ext_vector_type(16))) __bf16 v16bf;
typedef __attribute__((ext_vector_type(4)))  __bf16 v4bf;
typedef __attribute__((ext_vector_type(8)))  float  v8f;

// ---------------------------------------------------------------------------
// Fragment-order LDS swizzle (this code's contiguous k = 16*(lane>>4) + idx order, wave32):
//  A (16x32 MxK): lane = m + 16*(k>>4),     idx = k & 15
//  B (32x16 KxN): lane = n + 16*(k>>4),     idx = k & 15
// Each lane's 16-element fragment is contiguous (32 bytes) -> 2x ds_load_b128.
// ---------------------------------------------------------------------------
__device__ __forceinline__ v16bf frag_load(const __bf16* p) {
    return *(const v16bf*)p;              // 32B-aligned per-lane fragment
}

__device__ __forceinline__ v4bf cvt4(float4 v) {
    v4bf r;
    r[0] = (__bf16)v.x; r[1] = (__bf16)v.y;
    r[2] = (__bf16)v.z; r[3] = (__bf16)v.w;
    return r;
}

// ---------------------------------------------------------------------------
// Generic tiled WMMA GEMM:  C[M,N] = act(A[M,K] * B + bias)
//   transB=0 : B is [K,N] row-major  (C = A @ B)
//   transB=1 : B is [N,K] row-major  (C = A @ B^T)
// Block tile 64x64, BK=32, 128 threads = 4 waves (one 16-row strip each).
// LDS tiles are stored pre-swizzled in fragment order.
// ---------------------------------------------------------------------------
__global__ __launch_bounds__(128) void wmma_gemm_kernel(
    const float* __restrict__ A, int lda,
    const float* __restrict__ B, int ldb, int transB,
    const float* __restrict__ bias, int relu,
    float* __restrict__ C, int M, int Nmat, int K)
{
    __shared__ __attribute__((aligned(32))) __bf16 sA[4 * 512]; // 4 row-strips
    __shared__ __attribute__((aligned(32))) __bf16 sB[4 * 512]; // 4 col-tiles

    const int row0 = blockIdx.x * 64;
    const int col0 = blockIdx.y * 64;
    const int t    = threadIdx.x;
    const int wave = t >> 5;
    const int lane = t & 31;

    v8f acc[4] = {};

    for (int k0 = 0; k0 < K; k0 += 32) {
        // ---- stage A (64 rows x 32 k): 1 row x 16 k per thread, float4 loads
        {
            int r   = t >> 1;
            int kcb = (t & 1) * 16;
            bool ok = (row0 + r) < M;
            const float* ap = A + (size_t)(row0 + r) * lda + k0 + kcb;
            int strip = r >> 4, m = r & 15;
#pragma unroll
            for (int c = 0; c < 4; ++c) {
                int k = kcb + c * 4;
                float4 v = ok ? *(const float4*)(ap + c * 4)
                              : make_float4(0.f, 0.f, 0.f, 0.f);
                int ln  = m + 16 * (k >> 4);
                int idx = k & 15;
                *(v4bf*)(sA + strip * 512 + ln * 16 + idx) = cvt4(v);
            }
        }
        // ---- stage B (32 k x 64 n)
        if (transB) {
            // B[j*ldb + k]: coalesced along k, vector path
            int jj  = t >> 1;
            int kcb = (t & 1) * 16;
            int jg  = col0 + jj;
            bool ok = jg < Nmat;
            const float* bp = B + (size_t)jg * ldb + k0 + kcb;
            int tile = jj >> 4, n = jj & 15;
#pragma unroll
            for (int c = 0; c < 4; ++c) {
                int k = kcb + c * 4;
                float4 v = ok ? *(const float4*)(bp + c * 4)
                              : make_float4(0.f, 0.f, 0.f, 0.f);
                int ln = n + 16 * (k >> 4);
                *(v4bf*)(sB + tile * 512 + ln * 16 + (k & 15)) = cvt4(v);
            }
        } else {
            // B[k*ldb + j]: coalesced along j, scalar swizzled stores
            int kk  = t >> 2;
            int jcb = (t & 3) * 16;
            const float* bp = B + (size_t)(k0 + kk) * ldb + col0 + jcb;
            int ln16 = 16 * (kk >> 4);
            int idx  = kk & 15;
#pragma unroll
            for (int i = 0; i < 16; ++i) {
                int j = jcb + i;
                float v = (col0 + j) < Nmat ? bp[i] : 0.f;
                sB[(j >> 4) * 512 + ((j & 15) + ln16) * 16 + idx] = (__bf16)v;
            }
        }
        __syncthreads();

        v16bf aF = frag_load(sA + wave * 512 + lane * 16);
#pragma unroll
        for (int tt = 0; tt < 4; ++tt) {
            v16bf bF = frag_load(sB + tt * 512 + lane * 16);
            acc[tt] = __builtin_amdgcn_wmma_f32_16x16x32_bf16(
                false, aF, false, bF, (short)0, acc[tt], false, false);
        }
        __syncthreads();
    }

    // epilogue: D layout — col = lane&15, row = v + 8*(lane>>4)
    const int n  = lane & 15;
    const int mb = (lane >> 4) * 8;
#pragma unroll
    for (int tt = 0; tt < 4; ++tt) {
        int col = col0 + tt * 16 + n;
        if (col >= Nmat) continue;
        float bv = bias ? bias[col] : 0.0f;
#pragma unroll
        for (int v = 0; v < 8; ++v) {
            int row = row0 + wave * 16 + mb + v;
            if (row < M) {
                float x = acc[tt][v] + bv;
                if (relu) x = fmaxf(x, 0.0f);
                C[(size_t)row * Nmat + col] = x;
            }
        }
    }
}

// ---------------------------------------------------------------------------
// Edge scatter-add:  agg[dst[e], :] += m[src[e], :]   (H = 128)
// 32 lanes per edge, float4 gather (L2-resident), global_atomic_add_f32.
// ---------------------------------------------------------------------------
__global__ __launch_bounds__(256) void scatter_add_kernel(
    const float* __restrict__ m, const int* __restrict__ src,
    const int* __restrict__ dst, float* __restrict__ agg, int E)
{
    long long idx = (long long)blockIdx.x * 256 + threadIdx.x;
    long long e   = idx >> 5;
    if (e >= E) return;
    int q = (int)(idx & 31);
    int s = src[e];
    int d = dst[e];
    float4 v = reinterpret_cast<const float4*>(m + (size_t)s * 128)[q];
    float* out = agg + (size_t)d * 128 + q * 4;
    atomicAdd(out + 0, v.x);
    atomicAdd(out + 1, v.y);
    atomicAdd(out + 2, v.z);
    atomicAdd(out + 3, v.w);
}

// ---------------------------------------------------------------------------
// Fused GRU cell for a 32-row block:
//   gi = agg @ Wih^T (+bih),  gh = h @ Whh^T (+bhh)  (384 wide, kept in LDS)
//   r = sig(i_r+h_r); z = sig(i_z+h_z); n = tanh(i_n + r*h_n)
//   h' = (1-z)*n + z*h
// 256 threads = 8 waves: (wave&1) -> 16-row strip, (wave>>1) -> 96-col group.
// Dynamic LDS: sGI(48KB) + sGH(48KB) + sA(2KB) + sB(24KB) ~= 122KB.
// ---------------------------------------------------------------------------
__global__ __launch_bounds__(256) void gru_fused_kernel(
    const float* __restrict__ agg, const float* __restrict__ hin,
    const float* __restrict__ Wih, const float* __restrict__ Whh,
    const float* __restrict__ bih, const float* __restrict__ bhh,
    float* __restrict__ hout, int Nrows)
{
    extern __shared__ char smem[];
    float*  sGI = (float*)smem;                  // 32*384 f32
    float*  sGH = sGI + 32 * 384;                // 32*384 f32
    __bf16* sA  = (__bf16*)(sGH + 32 * 384);     // 2 strips * 512 (swizzled)
    __bf16* sB  = sA + 2 * 512;                  // 24 tiles * 512 (swizzled)

    const int t    = threadIdx.x;
    const int wave = t >> 5;
    const int lane = t & 31;
    const int row0 = blockIdx.x * 32;
    const int wr   = wave & 1;    // row strip (0/1)
    const int wc   = wave >> 1;   // col group (0..3) * 96

#pragma unroll
    for (int p = 0; p < 2; ++p) {
        const float* Amat = p ? hin : agg;
        const float* Wmat = p ? Whh : Wih;        // [384,128] row-major (W^T use)
        float*       sOut = p ? sGH : sGI;

        v8f acc[6] = {};

        for (int k0 = 0; k0 < 128; k0 += 32) {
            // stage A (32x32): one float4 chunk per thread
            {
                int r  = t >> 3;
                int kc = (t & 7) * 4;
                bool ok = (row0 + r) < Nrows;
                const float* ap = Amat + (size_t)(row0 + r) * 128 + k0 + kc;
                float4 v = ok ? *(const float4*)ap : make_float4(0.f, 0.f, 0.f, 0.f);
                int strip = r >> 4, m = r & 15;
                int ln  = m + 16 * (kc >> 4);
                int idx = kc & 15;
                *(v4bf*)(sA + strip * 512 + ln * 16 + idx) = cvt4(v);
            }
            // stage B (32 k x 384 j): B(k,j) = W[j*128+k]; coalesced float4
#pragma unroll
            for (int it = 0; it < 12; ++it) {
                int c  = t + 256 * it;            // 3072 chunks total
                int j  = c >> 3;
                int kc = (c & 7) * 4;
                float4 v = *(const float4*)(Wmat + (size_t)j * 128 + k0 + kc);
                int tile = j >> 4, n = j & 15;
                int ln = n + 16 * (kc >> 4);
                *(v4bf*)(sB + tile * 512 + ln * 16 + (kc & 15)) = cvt4(v);
            }
            __syncthreads();

            v16bf aF = frag_load(sA + wr * 512 + lane * 16);
#pragma unroll
            for (int tt = 0; tt < 6; ++tt) {
                v16bf bF = frag_load(sB + (wc * 6 + tt) * 512 + lane * 16);
                acc[tt] = __builtin_amdgcn_wmma_f32_16x16x32_bf16(
                    false, aF, false, bF, (short)0, acc[tt], false, false);
            }
            __syncthreads();
        }

        // dump accumulators to LDS gate buffer
        const int n  = lane & 15;
        const int mb = (lane >> 4) * 8;
#pragma unroll
        for (int tt = 0; tt < 6; ++tt) {
            int col = wc * 96 + tt * 16 + n;
#pragma unroll
            for (int v = 0; v < 8; ++v) {
                int r = wr * 16 + mb + v;
                sOut[r * 384 + col] = acc[tt][v];
            }
        }
        __syncthreads();
    }

    // gate math: 32 rows x 128 cols
    for (int i = t; i < 32 * 128; i += 256) {
        int r  = i >> 7;
        int j  = i & 127;
        int gr = row0 + r;
        if (gr >= Nrows) continue;
        float ir  = sGI[r * 384 + j]       + bih[j];
        float iz  = sGI[r * 384 + 128 + j] + bih[128 + j];
        float inn = sGI[r * 384 + 256 + j] + bih[256 + j];
        float hr  = sGH[r * 384 + j]       + bhh[j];
        float hz  = sGH[r * 384 + 128 + j] + bhh[128 + j];
        float hn  = sGH[r * 384 + 256 + j] + bhh[256 + j];
        float rg = 1.0f / (1.0f + __expf(-(ir + hr)));
        float zg = 1.0f / (1.0f + __expf(-(iz + hz)));
        float ng = tanhf(inn + rg * hn);
        float hv = hin[(size_t)gr * 128 + j];
        hout[(size_t)gr * 128 + j] = (1.0f - zg) * ng + zg * hv;
    }
}

// ---------------------------------------------------------------------------
extern "C" void kernel_launch(void* const* d_in, const int* in_sizes, int n_in,
                              void* d_out, int out_size, void* d_ws, size_t ws_size,
                              hipStream_t stream)
{
    const float* x     = (const float*)d_in[0];
    const int*   ei    = (const int*)d_in[1];
    const float* Win   = (const float*)d_in[2];
    const float* bin   = (const float*)d_in[3];
    const float* W1    = (const float*)d_in[4];
    const float* g1Wih = (const float*)d_in[5];
    const float* g1Whh = (const float*)d_in[6];
    const float* g1bih = (const float*)d_in[7];
    const float* g1bhh = (const float*)d_in[8];
    const float* W2    = (const float*)d_in[9];
    const float* g2Wih = (const float*)d_in[10];
    const float* g2Whh = (const float*)d_in[11];
    const float* g2bih = (const float*)d_in[12];
    const float* g2bhh = (const float*)d_in[13];
    float* out = (float*)d_out;

    const int N = 50000, E = 600000, DIN = 256, H = 128;
    const int* src = ei;
    const int* dst = ei + E;

    const size_t NH = (size_t)N * H;
    float* h0  = (float*)d_ws;     // input-MLP output / layer2 m-buffer
    float* t1  = h0 + NH;          // layer1 m-buffer, then layer1 h output
    float* agg = t1 + NH;          // edge aggregation (both layers)

    dim3 gemmGrid((N + 63) / 64, H / 64);
    const int scatterBlocks = (int)(((long long)E * 32 + 255) / 256);
    const int gruBlocks     = (N + 31) / 32;
    const size_t gruLds = (size_t)(2 * 32 * 384) * sizeof(float)
                        + (size_t)(2 * 512 + 24 * 512) * sizeof(__bf16);
    hipFuncSetAttribute((const void*)gru_fused_kernel,
                        hipFuncAttributeMaxDynamicSharedMemorySize, (int)gruLds);

    // 1) h0 = relu(x @ Win^T + bin)
    wmma_gemm_kernel<<<gemmGrid, 128, 0, stream>>>(
        x, DIN, Win, DIN, /*transB=*/1, bin, /*relu=*/1, h0, N, H, DIN);

    // ---- layer 1: hin = h0, hout = t1 (m aliases t1: dead before GRU writes)
    wmma_gemm_kernel<<<gemmGrid, 128, 0, stream>>>(
        h0, H, W1, H, /*transB=*/0, nullptr, /*relu=*/0, t1, N, H, H);
    hipMemsetAsync(agg, 0, NH * sizeof(float), stream);
    scatter_add_kernel<<<scatterBlocks, 256, 0, stream>>>(t1, src, dst, agg, E);
    gru_fused_kernel<<<gruBlocks, 256, gruLds, stream>>>(
        agg, h0, g1Wih, g1Whh, g1bih, g1bhh, t1, N);

    // ---- layer 2: hin = t1, hout = d_out (m reuses h0, which is dead)
    wmma_gemm_kernel<<<gemmGrid, 128, 0, stream>>>(
        t1, H, W2, H, /*transB=*/0, nullptr, /*relu=*/0, h0, N, H, H);
    hipMemsetAsync(agg, 0, NH * sizeof(float), stream);
    scatter_add_kernel<<<scatterBlocks, 256, 0, stream>>>(h0, src, dst, agg, E);
    gru_fused_kernel<<<gruBlocks, 256, gruLds, stream>>>(
        agg, t1, g2Wih, g2Whh, g2bih, g2bhh, out, N);
}
